// SpGraphAttentionLayer_17918603559514
// MI455X (gfx1250) — compile-verified, matrix operand order checked by reading, not run on hardware
//
#include <hip/hip_runtime.h>

// ---------------- problem constants ----------------
constexpr int   NN    = 50000;
constexpr int   INF_  = 128;
constexpr int   OUTF  = 128;
constexpr int   NREL  = 128;
constexpr int   E1C   = 400000;
constexpr int   E2C   = 100000;
constexpr int   ET    = E1C + E2C;      // 500000 edges total
constexpr int   KTOT  = 2 * INF_ + NREL; // 384
constexpr int   NKS   = KTOT / 32;       // 12 K-steps of 32
constexpr int   NNB   = OUTF / 16;       // 8 N-blocks of 16
constexpr int   NTILES = ET / 16;        // 31250 edge tiles

#define ALPHA 0.2f
#define DAMP  0.85f
#define EPSV  1e-12f

typedef __attribute__((ext_vector_type(16))) __bf16 v16bf;
typedef __attribute__((ext_vector_type(8)))  float  v8f;

__device__ __forceinline__ float lrelu(float x) { return x >= 0.f ? x : ALPHA * x; }

// ---------------- utility: zero fill ----------------
__global__ void zero_kernel(float* __restrict__ p, long n) {
    long i = (long)blockIdx.x * blockDim.x + threadIdx.x;
    long s = (long)gridDim.x * blockDim.x;
    for (; i < n; i += s) p[i] = 0.f;
}

// ---------------- prep: swizzle weight a[OUTF][KTOT] (f32) into per-lane
// v16bf B-tile layout: [ks][nb][lane][16 bf16].
// lane<16 holds col n=nb*16+lane, k in {kb..kb+7, kb+16..kb+23}, kb=ks*32
// lane>=16 same col set shifted: kb+=8 (mirrors the 16-bit A-matrix layout). ----
__global__ void prep_b_kernel(const float* __restrict__ a, __bf16* __restrict__ bswz) {
    int t = blockIdx.x * blockDim.x + threadIdx.x;
    if (t >= NKS * NNB * 32) return;
    int lane = t & 31;
    int nb   = (t >> 5) & (NNB - 1);
    int ks   = t >> 8;
    int hi = lane >> 4, lo = lane & 15;
    int n  = nb * 16 + lo;
    int kb = ks * 32 + hi * 8;
    __bf16* o = bswz + (size_t)t * 16;
#pragma unroll
    for (int j = 0; j < 8; ++j) o[j]     = (__bf16)a[n * KTOT + kb + j];
#pragma unroll
    for (int j = 0; j < 8; ++j) o[8 + j] = (__bf16)a[n * KTOT + kb + 16 + j];
}

// ---------------- K1: edge GEMM (WMMA bf16) + leakyrelu + a_2 dot + edge_e ----
__global__ __launch_bounds__(256)
void edge_gemm_kernel(const float* __restrict__ input,
                      const int*   __restrict__ edge,      // [2, E1]
                      const float* __restrict__ eemb,      // [E1, NREL]
                      const int*   __restrict__ enh,       // [2, E2]
                      const float* __restrict__ eembnh,    // [E2, NREL]
                      const float* __restrict__ a2,        // [OUTF]
                      const __bf16* __restrict__ bswz,
                      float* __restrict__ edge_m,          // [ET, OUTF]
                      float* __restrict__ edge_e,          // [ET]
                      float* __restrict__ a_rowsum)        // [NN]
{
    __shared__ __bf16 sB[NKS * NNB * 32 * 16];   // 98304 bytes
    // stage swizzled B into LDS (16B chunks)
    {
        const uint4* g = (const uint4*)bswz;
        uint4* l = (uint4*)sB;
        const int nchunk = NKS * NNB * 32 * 16 * 2 / 16;   // 6144
        for (int i = threadIdx.x; i < nchunk; i += blockDim.x) l[i] = g[i];
    }
    __syncthreads();

    const int lane = threadIdx.x & 31;
    const int wave = threadIdx.x >> 5;
    const int hi = lane >> 4, lo = lane & 15;
    const int wstride = gridDim.x * (blockDim.x >> 5);
    const v16bf* sBv = (const v16bf*)sB;

    for (int tile = blockIdx.x * (blockDim.x >> 5) + wave; tile < NTILES; tile += wstride) {
        // this lane gathers row m = lo of the 16-edge tile (both lane halves do)
        const int e = tile * 16 + lo;
        int s, d;
        const float* relrow;
        if (e < E1C) {
            s = edge[e]; d = edge[E1C + e];
            relrow = eemb + (size_t)e * NREL;
        } else {
            int e2 = e - E1C;
            s = enh[e2]; d = enh[E2C + e2];
            relrow = eembnh + (size_t)e2 * NREL;
        }
        const float* seg0 = input + (size_t)s * INF_;
        const float* seg1 = input + (size_t)d * INF_;

        v8f acc[NNB];
#pragma unroll
        for (int nb = 0; nb < NNB; ++nb) acc[nb] = (v8f)(0.f);

#pragma unroll
        for (int ks = 0; ks < NKS; ++ks) {
            const float* base = (ks < 4) ? seg0 : (ks < 8) ? seg1 : relrow;
            const int kk = (ks & 3) * 32 + hi * 8;
            float4 f0 = *(const float4*)(base + kk);
            float4 f1 = *(const float4*)(base + kk + 4);
            float4 f2 = *(const float4*)(base + kk + 16);
            float4 f3 = *(const float4*)(base + kk + 20);
            v16bf A;
            A[0]  = (__bf16)f0.x; A[1]  = (__bf16)f0.y; A[2]  = (__bf16)f0.z; A[3]  = (__bf16)f0.w;
            A[4]  = (__bf16)f1.x; A[5]  = (__bf16)f1.y; A[6]  = (__bf16)f1.z; A[7]  = (__bf16)f1.w;
            A[8]  = (__bf16)f2.x; A[9]  = (__bf16)f2.y; A[10] = (__bf16)f2.z; A[11] = (__bf16)f2.w;
            A[12] = (__bf16)f3.x; A[13] = (__bf16)f3.y; A[14] = (__bf16)f3.z; A[15] = (__bf16)f3.w;
#pragma unroll
            for (int nb = 0; nb < NNB; ++nb) {
                v16bf B = sBv[(ks * NNB + nb) * 32 + lane];
                acc[nb] = __builtin_amdgcn_wmma_f32_16x16x32_bf16(
                    false, A, false, B, (short)0, acc[nb], false, false);
            }
        }

        // epilogue: leakyrelu, store edge_m, accumulate a_2 dot partials.
        // C/D layout: acc[nb][r] = edge_m[m = r + 8*hi][n = nb*16 + lo]
        float p[8];
#pragma unroll
        for (int r = 0; r < 8; ++r) p[r] = 0.f;
#pragma unroll
        for (int nb = 0; nb < NNB; ++nb) {
            const int n = nb * 16 + lo;
            const float a2n = a2[n];
#pragma unroll
            for (int r = 0; r < 8; ++r) {
                float x = lrelu(acc[nb][r]);
                const int m = r + 8 * hi;
                edge_m[(size_t)(tile * 16 + m) * OUTF + n] = x;
                p[r] += x * a2n;
            }
        }
        // butterfly-reduce each p[r] over the 16 lanes of this half-wave
#pragma unroll
        for (int r = 0; r < 8; ++r) {
#pragma unroll
            for (int mask = 1; mask < 16; mask <<= 1)
                p[r] += __shfl_xor(p[r], mask, 32);
        }
        // lanes 0..7 own rows 0..7 (hi=0), lanes 24..31 own rows 8..15 (hi=1)
        const bool active = (hi == 0) ? (lo < 8) : (lo >= 8);
        if (active) {
            const int r = lo & 7;
            float sv = p[0];
#pragma unroll
            for (int rr = 1; rr < 8; ++rr) if (r == rr) sv = p[rr];
            float pw = -lrelu(sv);
            float ee = __expf(pw);
            edge_e[e] = ee;
            unsafeAtomicAdd(&a_rowsum[d], ee);
        }
    }
}

// ---------------- K2: rel_att = edge_e / a_rowsum[dst]; e_rowsum[src] += ----
__global__ void rel_att_kernel(const float* __restrict__ edge_e,
                               const int* __restrict__ edge, const int* __restrict__ enh,
                               const float* __restrict__ a_rowsum,
                               float* __restrict__ rel_att, float* __restrict__ e_rowsum) {
    int e = blockIdx.x * blockDim.x + threadIdx.x;
    if (e >= ET) return;
    int s, d;
    if (e < E1C) { s = edge[e]; d = edge[E1C + e]; }
    else { int e2 = e - E1C; s = enh[e2]; d = enh[E2C + e2]; }
    float rs = a_rowsum[d];
    rs = (rs == 0.f) ? EPSV : rs;
    float ra = edge_e[e] / rs;
    rel_att[e] = ra;
    unsafeAtomicAdd(&e_rowsum[s], ra);
}

// ---------------- K3: val scatter into e_colsum ----------------
__global__ void val_kernel(const float* __restrict__ rel_att,
                           const int* __restrict__ edge, const int* __restrict__ enh,
                           const float* __restrict__ entity_rank,
                           const float* __restrict__ e_rowsum,
                           float* __restrict__ e_colsum) {
    int e = blockIdx.x * blockDim.x + threadIdx.x;
    if (e >= ET) return;
    int s, d;
    if (e < E1C) { s = edge[e]; d = edge[E1C + e]; }
    else { int e2 = e - E1C; s = enh[e2]; d = enh[E2C + e2]; }
    float er = e_rowsum[s];
    er = (er == 0.f) ? EPSV : er;
    float v = rel_att[e] * entity_rank[s] / er;
    unsafeAtomicAdd(&e_colsum[d], v);
}

// ---------------- K4: new_rank ----------------
__global__ void rank_kernel(const float* __restrict__ e_colsum, float* __restrict__ out_rank) {
    int i = blockIdx.x * blockDim.x + threadIdx.x;
    if (i >= NN) return;
    out_rank[i] = (1.f - DAMP) + DAMP * e_colsum[i];
}

// ---------------- K5: rank-weighted propagation (one wave per edge) ----------
__global__ __launch_bounds__(256)
void prop_kernel(const float* __restrict__ rel_att,
                 const int* __restrict__ edge, const int* __restrict__ enh,
                 const float* __restrict__ new_rank,
                 const float* __restrict__ edge_m,
                 float* __restrict__ h_acc) {
    int gtid = blockIdx.x * blockDim.x + threadIdx.x;
    int e = gtid >> 5;
    int lane = gtid & 31;
    if (e >= ET) return;
    int s, d;
    if (e < E1C) { s = edge[e]; d = edge[E1C + e]; }
    else { int e2 = e - E1C; s = enh[e2]; d = enh[E2C + e2]; }
    float w = rel_att[e] * new_rank[s];
    float4 m4 = ((const float4*)(edge_m + (size_t)e * OUTF))[lane];
    float* hp = h_acc + (size_t)d * OUTF + lane * 4;
    unsafeAtomicAdd(hp + 0, w * m4.x);
    unsafeAtomicAdd(hp + 1, w * m4.y);
    unsafeAtomicAdd(hp + 2, w * m4.z);
    unsafeAtomicAdd(hp + 3, w * m4.w);
}

// ---------------- K6: in-place ELU on h ----------------
__global__ void elu_kernel(float* __restrict__ h, long n) {
    long i = (long)blockIdx.x * blockDim.x + threadIdx.x;
    if (i >= n) return;
    float x = h[i];
    h[i] = (x > 0.f) ? x : (__expf(x) - 1.f);
}

// ---------------- launch ----------------
extern "C" void kernel_launch(void* const* d_in, const int* in_sizes, int n_in,
                              void* d_out, int out_size, void* d_ws, size_t ws_size,
                              hipStream_t stream) {
    const float* input  = (const float*)d_in[0];
    const int*   edge   = (const int*)d_in[1];
    const float* eemb   = (const float*)d_in[2];
    const int*   enh    = (const int*)d_in[3];
    const float* eembnh = (const float*)d_in[4];
    const float* erank  = (const float*)d_in[5];
    const float* a      = (const float*)d_in[6];
    const float* a2     = (const float*)d_in[7];

    char* ws = (char*)d_ws;
    auto al = [](size_t x) { return (x + 255) & ~(size_t)255; };
    size_t o_b  = 0;
    size_t o_m  = al(o_b + (size_t)NKS * NNB * 32 * 16 * 2);  // bswz: 96 KB
    size_t o_ee = al(o_m + (size_t)ET * OUTF * 4);            // edge_m: 256 MB
    size_t o_ra = al(o_ee + (size_t)ET * 4);                  // edge_e: 2 MB
    size_t o_ar = al(o_ra + (size_t)ET * 4);                  // rel_att: 2 MB
    size_t o_er = al(o_ar + (size_t)NN * 4);
    size_t o_ec = al(o_er + (size_t)NN * 4);

    __bf16* bswz   = (__bf16*)(ws + o_b);
    float* edge_m  = (float*)(ws + o_m);
    float* edge_e  = (float*)(ws + o_ee);
    float* rel_att = (float*)(ws + o_ra);
    float* a_row   = (float*)(ws + o_ar);
    float* e_row   = (float*)(ws + o_er);
    float* e_col   = (float*)(ws + o_ec);

    float* h_out    = (float*)d_out;             // [NN, OUTF]
    float* rank_out = h_out + (size_t)NN * OUTF; // [NN]

    zero_kernel<<<2048, 256, 0, stream>>>(h_out, (long)NN * OUTF);
    zero_kernel<<<64, 256, 0, stream>>>(a_row, NN);
    zero_kernel<<<64, 256, 0, stream>>>(e_row, NN);
    zero_kernel<<<64, 256, 0, stream>>>(e_col, NN);

    prep_b_kernel<<<(NKS * NNB * 32 + 255) / 256, 256, 0, stream>>>(a, bswz);

    edge_gemm_kernel<<<512, 256, 0, stream>>>(input, edge, eemb, enh, eembnh, a2,
                                              bswz, edge_m, edge_e, a_row);

    rel_att_kernel<<<(ET + 255) / 256, 256, 0, stream>>>(edge_e, edge, enh, a_row,
                                                         rel_att, e_row);

    val_kernel<<<(ET + 255) / 256, 256, 0, stream>>>(rel_att, edge, enh, erank,
                                                     e_row, e_col);

    rank_kernel<<<(NN + 255) / 256, 256, 0, stream>>>(e_col, rank_out);

    prop_kernel<<<(ET * 32) / 256, 256, 0, stream>>>(rel_att, edge, enh, rank_out,
                                                     edge_m, h_out);

    elu_kernel<<<((long)NN * OUTF + 255) / 256, 256, 0, stream>>>(h_out, (long)NN * OUTF);
}
